// semantic_40346922778649
// MI455X (gfx1250) — matrix operand order repaired and checked by reading, not compiled, not run on hardware
//
#include <hip/hip_runtime.h>
#include <stdint.h>

typedef __attribute__((ext_vector_type(2)))  float  v2f;
typedef __attribute__((ext_vector_type(4)))  float  v4f;
typedef __attribute__((ext_vector_type(8)))  float  v8f;
typedef __attribute__((ext_vector_type(16))) __bf16 v16bf;

typedef __attribute__((address_space(1))) int as1_int;
typedef __attribute__((address_space(3))) int as3_int;

#if defined(__has_builtin)
#if __has_builtin(__builtin_amdgcn_tanhf)
#define HAVE_AMDGCN_TANH 1
#endif
#if __has_builtin(__builtin_amdgcn_global_load_async_to_lds_b32)
#define HAVE_ASYNC_LDS 1
#endif
#endif

__device__ __forceinline__ float fast_tanh(float x) {
#ifdef HAVE_AMDGCN_TANH
  return __builtin_amdgcn_tanhf(x);
#else
  return tanhf(x);
#endif
}

#ifdef HAVE_ASYNC_LDS
__device__ __forceinline__ void wait_async0() {
#if __has_builtin(__builtin_amdgcn_s_wait_asynccnt)
  __builtin_amdgcn_s_wait_asynccnt(0);
#else
  asm volatile("s_wait_asynccnt 0" ::: "memory");
#endif
}
#endif

// One dword global -> LDS (async on CDNA5; sync fallback elsewhere)
__device__ __forceinline__ void copy_b32_to_lds(const float* g, float* l) {
#ifdef HAVE_ASYNC_LDS
  __builtin_amdgcn_global_load_async_to_lds_b32((as1_int*)g, (as3_int*)l, 0, 0);
#else
  *l = *g;
#endif
}

// ---------------------------------------------------------------------------
// fd[k,d] = sum_w word[k,w] * W2[d,w]       (K=80, W=300)
// ---------------------------------------------------------------------------
__global__ __launch_bounds__(256) void fd_kernel(const float* __restrict__ word,
                                                 const float* __restrict__ W2,
                                                 float* __restrict__ fd, int D) {
  int idx = blockIdx.x * blockDim.x + threadIdx.x;
  if (idx >= 80 * D) return;
  int k = idx / D, d = idx - k * D;
  const float* wr = word + (size_t)k * 300;
  const float* w2 = W2 + (size_t)d * 300;
  float acc = 0.f;
  for (int w = 0; w < 300; ++w) acc += wr[w] * w2[w];
  fd[idx] = acc;
}

// ---------------------------------------------------------------------------
// v[d] = sum_e Wa[e] * W3[e,d]   (Wa folded through W3; b3/ba shift every
// score by the same constant, which cancels in the spatial softmax)
// ---------------------------------------------------------------------------
__global__ __launch_bounds__(256) void v_kernel(const float* __restrict__ Wa,
                                                const float* __restrict__ W3,
                                                float* __restrict__ v, int D) {
  int d = blockIdx.x * blockDim.x + threadIdx.x;
  if (d >= D) return;
  float acc = 0.f;
  for (int e = 0; e < D; ++e) acc += Wa[e] * W3[(size_t)e * D + d];
  v[d] = acc;
}

// ---------------------------------------------------------------------------
// fw[m, n] = sum_c fmap[b, c, hw] * W1[n, c],  m = b*HW + hw
// bf16 WMMA 16x16x32 w/ f32 accumulate. A tile (16 rows x 32 K) staged once
// per block into LDS via async global->LDS; 8 waves = 8 adjacent n-tiles.
// LDS layout [c][row] with stride 17 to decouple the two half-wave groups.
// ---------------------------------------------------------------------------
__global__ __launch_bounds__(256) void gemm1_bf16(const float* __restrict__ fmap, // [B,C,HW]
                                                  const float* __restrict__ W1,   // [D,C]
                                                  float* __restrict__ fw,         // [M,D]
                                                  int C, int HW, int D) {
  __shared__ float shA[32 * 17];
  const int lane = threadIdx.x & 31;
  const int wave = threadIdx.x >> 5;
  const int half = lane >> 4;       // 0 | 1
  const int l16  = lane & 15;
  const int m0   = blockIdx.y * 16;
  const int n0   = (blockIdx.x * 8 + wave) * 16;

  // cooperative A-tile loader mapping: thread t -> (row = t&15, c = t>>4, c+16)
  const int lrow = threadIdx.x & 15;
  const int lc   = threadIdx.x >> 4;          // 0..15
  const int lm   = m0 + lrow;
  const int lb   = lm / HW;
  const int lhw  = lm - lb * HW;
  const float* aBase = fmap + (size_t)lb * C * HW + lhw;   // + c*HW

  const int ncol = n0 + l16;
  const float* bBase = W1 + (size_t)ncol * C;              // + c
  const int akb = half * 8;
  const int bkb = half * 8;

  v8f acc = {};
  for (int c0 = 0; c0 < C; c0 += 32) {
    // stage A tile: 512 dwords, 2 per thread (async -> LDS on CDNA5)
    copy_b32_to_lds(aBase + (size_t)(c0 + lc) * HW,      &shA[lc * 17 + lrow]);
    copy_b32_to_lds(aBase + (size_t)(c0 + lc + 16) * HW, &shA[(lc + 16) * 17 + lrow]);
#ifdef HAVE_ASYNC_LDS
    wait_async0();
#endif
    __syncthreads();

    // A fragment (16-bit A 16x32 layout): lane<16 -> K 0..7,16..23 of row l16
    v16bf av, bv;
#pragma unroll
    for (int j = 0; j < 8; ++j) {
      av[j]     = (__bf16)shA[(akb + j) * 17 + l16];
      av[8 + j] = (__bf16)shA[(akb + 16 + j) * 17 + l16];
    }
    // B fragment: lane<16 -> K 0..7,16..23 of W1 row `ncol`
    const float* bp = bBase + c0 + bkb;
    v4f b0 = *(const v4f*)(bp + 0);
    v4f b1 = *(const v4f*)(bp + 4);
    v4f b2 = *(const v4f*)(bp + 16);
    v4f b3 = *(const v4f*)(bp + 20);
    __builtin_prefetch(bp + 32, 0, 3);       // -> global_prefetch_b8
#pragma unroll
    for (int j = 0; j < 4; ++j) {
      bv[j]      = (__bf16)b0[j];
      bv[4 + j]  = (__bf16)b1[j];
      bv[8 + j]  = (__bf16)b2[j];
      bv[12 + j] = (__bf16)b3[j];
    }
    acc = __builtin_amdgcn_wmma_f32_16x16x32_bf16(false, av, false, bv,
                                                  (short)0, acc, false, false);
    __syncthreads();   // reads done before next iteration overwrites shA
  }
  // C/D layout: VGPR r -> M = r + 8*half, N = lane&15
#pragma unroll
  for (int r = 0; r < 8; ++r) {
    int m = m0 + r + half * 8;
    fw[(size_t)m * D + ncol] = acc[r];
  }
}

// ---------------------------------------------------------------------------
// score[b,k,hw] = sum_d tanh(fw[m,d] * fd[k,d]) * v[d]
// 8 spatial positions per block (fw rows staged in LDS) so each fd element
// fetched from L2 is reused 8x; wave w handles classes w, w+8, ...
// ---------------------------------------------------------------------------
#define SCORE_P 8
__global__ __launch_bounds__(256) void score_kernel(const float* __restrict__ fw,  // [M,D]
                                                    const float* __restrict__ fd,  // [80,D]
                                                    const float* __restrict__ v,   // [D]
                                                    float* __restrict__ score,     // [B,80,HW]
                                                    int D, int HW) {
  __shared__ float sh[SCORE_P * 1024 + 1024];   // 8 fw rows + v (D<=1024)
  float* vL = sh + SCORE_P * D;
  const int p0 = blockIdx.x * SCORE_P;
#pragma unroll
  for (int pp = 0; pp < SCORE_P; ++pp)
    for (int i = threadIdx.x; i < D; i += 256)
      sh[pp * D + i] = fw[(size_t)(p0 + pp) * D + i];
  for (int i = threadIdx.x; i < D; i += 256) vL[i] = v[i];
  __syncthreads();

  const int lane = threadIdx.x & 31;
  const int wave = threadIdx.x >> 5;
  for (int k = wave; k < 80; k += 8) {
    const float* fdk = fd + (size_t)k * D;
    float acc[SCORE_P];
#pragma unroll
    for (int pp = 0; pp < SCORE_P; ++pp) acc[pp] = 0.f;
    for (int i = lane; i < D; i += 32) {
      float fdv = fdk[i];
      float vv  = vL[i];
#pragma unroll
      for (int pp = 0; pp < SCORE_P; ++pp)
        acc[pp] += fast_tanh(sh[pp * D + i] * fdv) * vv;
    }
#pragma unroll
    for (int pp = 0; pp < SCORE_P; ++pp) {
      float a = acc[pp];
#pragma unroll
      for (int off = 16; off >= 1; off >>= 1)
        a += __shfl_xor(a, off, 32);
      if (lane == 0) {
        int p  = p0 + pp;
        int b  = p / HW;
        int hw = p - b * HW;
        score[((size_t)b * 80 + k) * HW + hw] = a;
      }
    }
  }
}

// ---------------------------------------------------------------------------
// In-place softmax over the HW axis, one block per (b,k) row.
// ---------------------------------------------------------------------------
__global__ __launch_bounds__(256) void softmax_kernel(float* __restrict__ s, int HW) {
  float* row = s + (size_t)blockIdx.x * HW;
  __shared__ float red[8];
  const int lane = threadIdx.x & 31, wave = threadIdx.x >> 5;

  float mx = -3.4e38f;
  for (int i = threadIdx.x; i < HW; i += blockDim.x) mx = fmaxf(mx, row[i]);
#pragma unroll
  for (int off = 16; off >= 1; off >>= 1) mx = fmaxf(mx, __shfl_xor(mx, off, 32));
  if (lane == 0) red[wave] = mx;
  __syncthreads();
  mx = fmaxf(fmaxf(fmaxf(red[0], red[1]), fmaxf(red[2], red[3])),
             fmaxf(fmaxf(red[4], red[5]), fmaxf(red[6], red[7])));
  __syncthreads();

  float sum = 0.f;
  for (int i = threadIdx.x; i < HW; i += blockDim.x) sum += __expf(row[i] - mx);
#pragma unroll
  for (int off = 16; off >= 1; off >>= 1) sum += __shfl_xor(sum, off, 32);
  if (lane == 0) red[wave] = sum;
  __syncthreads();
  sum = ((red[0] + red[1]) + (red[2] + red[3])) + ((red[4] + red[5]) + (red[6] + red[7]));
  float inv = 1.f / sum;
  for (int i = threadIdx.x; i < HW; i += blockDim.x)
    row[i] = __expf(row[i] - mx) * inv;
}

// ---------------------------------------------------------------------------
// out[b,k,c] = sum_hw coeff[b,k,hw] * fmap[b,c,hw]
// Exact-precision f32 WMMA (V_WMMA_F32_16X16X4_F32); both operands are
// contiguous along the reduction (hw) axis.
// ---------------------------------------------------------------------------
__global__ __launch_bounds__(256) void pool_wmma(const float* __restrict__ coeff, // [B,80,HW]
                                                 const float* __restrict__ fmap,  // [B,C,HW]
                                                 float* __restrict__ out,         // [B,80,C]
                                                 int C, int HW) {
  const int lane = threadIdx.x & 31;
  const int wave = threadIdx.x >> 5;
  const int half = lane >> 4;
  const int l16  = lane & 15;
  const int b  = blockIdx.z;
  const int m0 = blockIdx.y * 16;                 // class tile (80 = 5*16)
  const int n0 = (blockIdx.x * 8 + wave) * 16;    // channel tile

  const float* aRow = coeff + ((size_t)b * 80 + m0 + l16) * HW;
  const float* bRow = fmap + ((size_t)b * C + n0 + l16) * HW;

  v8f acc = {};
  for (int kk = 0; kk < HW; kk += 4) {
    v2f a  = *(const v2f*)(aRow + kk + 2 * half);
    v2f bb = *(const v2f*)(bRow + kk + 2 * half);
    acc = __builtin_amdgcn_wmma_f32_16x16x4_f32(false, a, false, bb,
                                                (short)0, acc, false, false);
  }
#pragma unroll
  for (int r = 0; r < 8; ++r) {
    int k = m0 + r + half * 8;
    out[((size_t)b * 80 + k) * C + n0 + l16] = acc[r];
  }
}

// ---------------------------------------------------------------------------
extern "C" void kernel_launch(void* const* d_in, const int* in_sizes, int n_in,
                              void* d_out, int out_size, void* d_ws, size_t ws_size,
                              hipStream_t stream) {
  (void)in_sizes; (void)n_in; (void)out_size; (void)ws_size;
  const float* f3   = (const float*)d_in[1];   // [4,1024,28,28]
  const float* f4   = (const float*)d_in[2];   // [4,2048,14,14]
  const float* word = (const float*)d_in[3];   // [80,300]
  const float* W1_3 = (const float*)d_in[4];
  const float* W2_3 = (const float*)d_in[5];
  const float* W3_3 = (const float*)d_in[6];
  const float* Wa_3 = (const float*)d_in[8];
  const float* W1_4 = (const float*)d_in[10];
  const float* W2_4 = (const float*)d_in[11];
  const float* W3_4 = (const float*)d_in[12];
  const float* Wa_4 = (const float*)d_in[14];
  float* out = (float*)d_out;

  const int B = 4, K = 80;
  const int C3 = 1024, HW3 = 784, D3 = 512,  M3 = B * HW3;  // stage 3
  const int C4 = 2048, HW4 = 196, D4 = 1024, M4 = B * HW4;  // stage 4

  float* ws  = (float*)d_ws;
  float* fw3 = ws;  ws += (size_t)M3 * D3;
  float* fw4 = ws;  ws += (size_t)M4 * D4;
  float* fd3 = ws;  ws += (size_t)K * D3;
  float* fd4 = ws;  ws += (size_t)K * D4;
  float* v3  = ws;  ws += D3;
  float* v4  = ws;  ws += D4;
  float* sc3 = ws;  ws += (size_t)B * K * HW3;
  float* sc4 = ws;  ws += (size_t)B * K * HW4;

  fd_kernel<<<(K * D3 + 255) / 256, 256, 0, stream>>>(word, W2_3, fd3, D3);
  fd_kernel<<<(K * D4 + 255) / 256, 256, 0, stream>>>(word, W2_4, fd4, D4);
  v_kernel<<<(D3 + 255) / 256, 256, 0, stream>>>(Wa_3, W3_3, v3, D3);
  v_kernel<<<(D4 + 255) / 256, 256, 0, stream>>>(Wa_4, W3_4, v4, D4);

  gemm1_bf16<<<dim3(D3 / 128, M3 / 16), 256, 0, stream>>>(f3, W1_3, fw3, C3, HW3, D3);
  gemm1_bf16<<<dim3(D4 / 128, M4 / 16), 256, 0, stream>>>(f4, W1_4, fw4, C4, HW4, D4);

  score_kernel<<<M3 / SCORE_P, 256, 0, stream>>>(fw3, fd3, v3, sc3, D3, HW3);
  score_kernel<<<M4 / SCORE_P, 256, 0, stream>>>(fw4, fd4, v4, sc4, D4, HW4);

  softmax_kernel<<<B * K, 256, 0, stream>>>(sc3, HW3);
  softmax_kernel<<<B * K, 256, 0, stream>>>(sc4, HW4);

  pool_wmma<<<dim3(C3 / 128, K / 16, B), 256, 0, stream>>>(sc3, f3, out, C3, HW3);
  pool_wmma<<<dim3(C4 / 128, K / 16, B), 256, 0, stream>>>(sc4, f4, out + (size_t)B * K * C3, C4, HW4);
}
